// TPRRNN_84232898609361
// MI455X (gfx1250) — compile-verified, matrix operand order checked by reading, not run on hardware
//
#include <hip/hip_runtime.h>

// ---------------------------------------------------------------------------
// TPR-RNN for MI455X (gfx1250, wave32).
//  - All matmuls via v_wmma_f32_16x16x32_bf16 (f32 accumulate).
//  - Weights pre-converted to bf16 once (single 4MB scratch, reused in stream
//    order) so GEMM inner loops have no weight-side cvt.
//  - Projection GEMM: 4 N-tiles per wave (A-fragment reused by 4 WMMAs).
//  - Recurrence: persistent kernel, static tile<->wave map (64 WGs x 8 waves
//    = 512 tiles); per-step A-tile staged in LDS (ds_load fragments);
//    device-wide atomic barrier between GEMM and attention phases.
// ---------------------------------------------------------------------------

#define T_MAX 512
#define B_DIM 64
#define D_DIM 512
#define H_DIM 1024
#define H2_DIM 2048
#define NR_DIM 64
#define NH_DIM 8
#define HD_DIM 128

typedef __attribute__((ext_vector_type(16))) __bf16 bf16x16;
typedef __attribute__((ext_vector_type(8)))  __bf16 bf16x8;
typedef __attribute__((ext_vector_type(8)))  float  v8f;

// --- WMMA fragment loaders (16x16x32 bf16, wave32 layouts per CDNA5 ISA) ---
// A (16x32): lane holds row (lane&15); lanes 0-15 carry K {0..7,16..23},
// lanes 16-31 carry K {8..15,24..31}. B (32x16) mirrored with col per lane.
__device__ __forceinline__ bf16x16 frag_from_f32(const float* rowbase, int k0, int half) {
  bf16x16 r;
#pragma unroll
  for (int j = 0; j < 8; ++j) {
    r[j]     = (__bf16)rowbase[k0 + half * 8 + j];
    r[j + 8] = (__bf16)rowbase[k0 + 16 + half * 8 + j];
  }
  return r;
}
// vectorized bf16 fragment load (two 16B loads; works for global or LDS ptrs)
__device__ __forceinline__ bf16x16 frag_from_bf16(const __bf16* rowbase, int k0, int half) {
  const bf16x8 lo = *(const bf16x8*)(rowbase + k0 + half * 8);
  const bf16x8 hi = *(const bf16x8*)(rowbase + k0 + 16 + half * 8);
  bf16x16 r;
#pragma unroll
  for (int j = 0; j < 8; ++j) { r[j] = lo[j]; r[j + 8] = hi[j]; }
  return r;
}
__device__ __forceinline__ bf16x16 frag_zero() {
  bf16x16 r;
#pragma unroll
  for (int j = 0; j < 16; ++j) r[j] = (__bf16)0.0f;
  return r;
}

// --- device-wide sense barrier (persistent kernel) ---
__device__ __forceinline__ void grid_barrier(unsigned* counter, unsigned* gen, unsigned nwg) {
  __builtin_amdgcn_fence(__ATOMIC_RELEASE, "agent");
  __syncthreads();
  if (threadIdx.x == 0) {
    unsigned g = __hip_atomic_load(gen, __ATOMIC_RELAXED, __HIP_MEMORY_SCOPE_AGENT);
    unsigned arrived =
        __hip_atomic_fetch_add(counter, 1u, __ATOMIC_ACQ_REL, __HIP_MEMORY_SCOPE_AGENT);
    if (arrived == nwg - 1u) {
      __hip_atomic_store(counter, 0u, __ATOMIC_RELAXED, __HIP_MEMORY_SCOPE_AGENT);
      __hip_atomic_store(gen, g + 1u, __ATOMIC_RELEASE, __HIP_MEMORY_SCOPE_AGENT);
    } else {
      while (__hip_atomic_load(gen, __ATOMIC_ACQUIRE, __HIP_MEMORY_SCOPE_AGENT) == g) {
        __builtin_amdgcn_s_sleep(8);
      }
    }
  }
  __syncthreads();
  __builtin_amdgcn_fence(__ATOMIC_ACQUIRE, "agent");
}

// --- batch_sizes / offsets from lengths (time-major packed layout) ---
__global__ void __launch_bounds__(256)
setup_offsets_kernel(const int* __restrict__ lengths, int* __restrict__ bsz,
                     int* __restrict__ offs) {
  for (int t = threadIdx.x; t < T_MAX; t += blockDim.x) {
    int c = 0;
    for (int b = 0; b < B_DIM; ++b) c += (lengths[b] > t) ? 1 : 0;
    bsz[t] = c;
  }
  __syncthreads();
  if (threadIdx.x == 0) {
    int acc = 0;
    for (int t = 0; t < T_MAX; ++t) { offs[t] = acc; acc += bsz[t]; }
  }
}

__global__ void __launch_bounds__(256)
f32_to_bf16_kernel(const float* __restrict__ src, __bf16* __restrict__ dst, int n) {
  for (int i = blockIdx.x * blockDim.x + threadIdx.x; i < n; i += gridDim.x * blockDim.x)
    dst[i] = (__bf16)src[i];
}

// --- C = A[M,K](f32) @ Wb[N,K]^T(bf16) + bias. One 16x64 strip per wave:
// A fragment loaded once per k-chunk and reused by 4 WMMAs.
__global__ void __launch_bounds__(256)
gemm_wmma_kernel(const float* __restrict__ A, const __bf16* __restrict__ Wb,
                 const float* __restrict__ bias, int M, int N, int K,
                 float* __restrict__ Cf, __bf16* __restrict__ Cb) {
  const int lane  = threadIdx.x & 31;
  const int half  = lane >> 4;
  const int wave  = blockIdx.x * (blockDim.x >> 5) + (threadIdx.x >> 5);
  const int nwave = gridDim.x * (blockDim.x >> 5);
  const int tm = (M + 15) >> 4, tn4 = N >> 6;  // strips of 4 N-tiles
  for (int strip = wave; strip < tm * tn4; strip += nwave) {
    const int mt = strip / tn4, nt0 = (strip % tn4) * 4;
    const int arow = mt * 16 + (lane & 15);
    const bool aok = arow < M;
    const float* arp = A + (size_t)arow * K;
    const __bf16* brp0 = Wb + (size_t)((nt0 + 0) * 16 + (lane & 15)) * K;
    const __bf16* brp1 = Wb + (size_t)((nt0 + 1) * 16 + (lane & 15)) * K;
    const __bf16* brp2 = Wb + (size_t)((nt0 + 2) * 16 + (lane & 15)) * K;
    const __bf16* brp3 = Wb + (size_t)((nt0 + 3) * 16 + (lane & 15)) * K;
    v8f c0 = {0.f,0.f,0.f,0.f,0.f,0.f,0.f,0.f};
    v8f c1 = c0, c2 = c0, c3 = c0;
    for (int k0 = 0; k0 < K; k0 += 32) {
      if (k0 + 128 < K) __builtin_prefetch(arp + k0 + 128, 0, 1);
      const bf16x16 a = aok ? frag_from_f32(arp, k0, half) : frag_zero();
      const bf16x16 b0 = frag_from_bf16(brp0, k0, half);
      const bf16x16 b1 = frag_from_bf16(brp1, k0, half);
      const bf16x16 b2 = frag_from_bf16(brp2, k0, half);
      const bf16x16 b3 = frag_from_bf16(brp3, k0, half);
      c0 = __builtin_amdgcn_wmma_f32_16x16x32_bf16(false, a, false, b0, (short)0, c0, false, false);
      c1 = __builtin_amdgcn_wmma_f32_16x16x32_bf16(false, a, false, b1, (short)0, c1, false, false);
      c2 = __builtin_amdgcn_wmma_f32_16x16x32_bf16(false, a, false, b2, (short)0, c2, false, false);
      c3 = __builtin_amdgcn_wmma_f32_16x16x32_bf16(false, a, false, b3, (short)0, c3, false, false);
    }
#pragma unroll
    for (int u = 0; u < 4; ++u) {
      const v8f c = (u == 0) ? c0 : (u == 1) ? c1 : (u == 2) ? c2 : c3;
      const int cc = (nt0 + u) * 16 + (lane & 15);
#pragma unroll
      for (int i = 0; i < 8; ++i) {
        const int r = mt * 16 + i + 8 * half;
        if (r < M) {
          float val = c[i] + bias[cc];
          if (Cf) Cf[(size_t)r * N + cc] = val;
          else    Cb[(size_t)r * N + cc] = (__bf16)val;
        }
      }
    }
  }
}

// --- persistent recurrence kernel (one layer) --------------------------------
// Static map: WG bid -> (mt = bid>>4, nt = (bid&15)*8 + waveId); 64 WGs x 8
// waves cover all 512 tiles. Per step, the WG stages its 16x1024 bf16 h-tile
// into LDS once; all 8 waves read A fragments via ds_load.
__global__ void __launch_bounds__(256)
tpru_recur_kernel(const __bf16* __restrict__ x2fg, const __bf16* __restrict__ Whb,
                  const float* __restrict__ bh, const float* __restrict__ v,
                  const int* __restrict__ lengths, const int* __restrict__ bsz,
                  const int* __restrict__ offs, float* __restrict__ h,
                  __bf16* __restrict__ hb, float* __restrict__ fg,
                  float* __restrict__ hs_out, float* __restrict__ hn_out,
                  unsigned* __restrict__ bar) {
  const int lane = threadIdx.x & 31;
  const int half = lane >> 4;
  const int wid  = threadIdx.x >> 5;
  const int gwave  = blockIdx.x * (blockDim.x >> 5) + wid;
  const int ngwave = gridDim.x * (blockDim.x >> 5);
  const int mt     = blockIdx.x >> 4;        // 0..3 (row tile of this WG)
  const int nt     = (blockIdx.x & 15) * 8 + wid;  // 0..127 (col tile of wave)

  __shared__ __bf16 lds_a[16 * H_DIM];       // 32KB staged h-tile
  __shared__ float  attn_lds[8][NR_DIM];     // 2KB attention weights

  // zero-init h state
  for (int i = blockIdx.x * blockDim.x + threadIdx.x; i < B_DIM * H_DIM;
       i += gridDim.x * blockDim.x) {
    h[i] = 0.f;
    hb[i] = (__bf16)0.f;
  }
  grid_barrier(bar, bar + 1, gridDim.x);

  for (int t = 0; t < T_MAX; ++t) {
    const int bs = bsz[t];
    const int off = offs[t];
    const int tm = (bs + 15) >> 4;  // 1..4 active row tiles
    // ---------------- phase 1: recurrent GEMM ----------------
    if (mt < tm) {  // WG-uniform condition
      // cooperative stage of this WG's 16x1024 A-tile (contiguous 32KB)
      const bf16x8* src = (const bf16x8*)(hb + (size_t)mt * 16 * H_DIM);
      bf16x8* dst = (bf16x8*)lds_a;
      for (int i = threadIdx.x; i < 16 * H_DIM / 8; i += blockDim.x) dst[i] = src[i];
      __syncthreads();

      const __bf16* arp = lds_a + (size_t)(lane & 15) * H_DIM;
      const __bf16* brp = Whb + (size_t)(nt * 16 + (lane & 15)) * H_DIM;
      v8f c = {0.f,0.f,0.f,0.f,0.f,0.f,0.f,0.f};
#pragma unroll 4
      for (int k0 = 0; k0 < H_DIM; k0 += 32) {
        const bf16x16 a = frag_from_bf16(arp, k0, half);
        const bf16x16 b = frag_from_bf16(brp, k0, half);
        c = __builtin_amdgcn_wmma_f32_16x16x32_bf16(false, a, false, b, (short)0, c,
                                                    false, false);
      }
#pragma unroll
      for (int i = 0; i < 8; ++i) {
        const int bb = mt * 16 + i + 8 * half;  // batch row (< 64 always)
        const int j  = nt * 16 + (lane & 15);
        float val = c[i] + bh[j];
        if (bb < bs) val += (float)x2fg[(size_t)(off + bb) * H2_DIM + j];
        fg[bb * H2_DIM + j] = val;
      }
    }
    grid_barrier(bar, bar + 1, gridDim.x);
    // ---------------- phase 2: attention + gated update ----------------
    {
      const int npairs = bs * NH_DIM;  // <= 512, one wave each
      for (int pair = gwave; pair < npairs; pair += ngwave) {
        const int bb = pair >> 3, hd = pair & 7;
        const float* fb = fg + bb * H2_DIM + hd * HD_DIM;           // f (bind query)
        const float* gb = fg + bb * H2_DIM + H_DIM + hd * HD_DIM;   // g (gate)
        // roles: lane handles roles {lane, lane+32}; dot over HD=128
        const float* vr0 = v + (size_t)lane * H2_DIM + hd * 2 * HD_DIM;
        const float* vr1 = v + (size_t)(lane + 32) * H2_DIM + hd * 2 * HD_DIM;
        float s0 = 0.f, s1 = 0.f;
        for (int d = 0; d < HD_DIM; ++d) {
          const float fv = fb[d];
          s0 = fmaf(fv, vr0[d], s0);
          s1 = fmaf(fv, vr1[d], s1);
        }
        // softmax over 64 roles (cross-lane reduce, wave32)
        float m = fmaxf(s0, s1);
        for (int o = 16; o > 0; o >>= 1) m = fmaxf(m, __shfl_xor(m, o, 32));
        const float e0 = __expf(s0 - m), e1 = __expf(s1 - m);
        float sum = e0 + e1;
        for (int o = 16; o > 0; o >>= 1) sum += __shfl_xor(sum, o, 32);
        const float inv = 1.f / sum;
        attn_lds[wid][lane]      = e0 * inv;
        attn_lds[wid][lane + 32] = e1 * inv;
        // bind: lane owns d = lane*4 .. lane*4+3
        float bind[4] = {0.f, 0.f, 0.f, 0.f};
        for (int rr = 0; rr < NR_DIM; ++rr) {
          const float arr = attn_lds[wid][rr];
          const float* ur = v + (size_t)rr * H2_DIM + hd * 2 * HD_DIM + HD_DIM;
#pragma unroll
          for (int q = 0; q < 4; ++q) bind[q] = fmaf(arr, ur[lane * 4 + q], bind[q]);
        }
        const int   hbase = bb * H_DIM + hd * HD_DIM + lane * 4;
        const size_t orow = (size_t)(off + bb) * H_DIM + hd * HD_DIM + lane * 4;
        const bool fin = (lengths[bb] == t + 1);
#pragma unroll
        for (int q = 0; q < 4; ++q) {
          const float g = gb[lane * 4 + q];
          const float gate = 1.f / (1.f + __expf(-g));
          const float hv = gate * tanhf(bind[q]) + (1.f - gate) * h[hbase + q];
          hs_out[orow + q] = hv;
          if (fin) hn_out[bb * H_DIM + hd * HD_DIM + lane * 4 + q] = hv;
          h[hbase + q]  = hv;
          hb[hbase + q] = (__bf16)hv;
        }
      }
    }
    grid_barrier(bar, bar + 1, gridDim.x);
  }
}

// ---------------------------------------------------------------------------
extern "C" void kernel_launch(void* const* d_in, const int* in_sizes, int n_in,
                              void* d_out, int out_size, void* d_ws, size_t ws_size,
                              hipStream_t stream) {
  const float* x       = (const float*)d_in[0];
  const int*   lengths = (const int*)d_in[2];
  const float* basis0  = (const float*)d_in[3];
  const float* basis1  = (const float*)d_in[4];
  const float* Wx0 = (const float*)d_in[5];   const float* bx0 = (const float*)d_in[6];
  const float* Wh0 = (const float*)d_in[7];   const float* bh0 = (const float*)d_in[8];
  const float* Wv0 = (const float*)d_in[9];   const float* bv0 = (const float*)d_in[10];
  const float* Wx1 = (const float*)d_in[11];  const float* bx1 = (const float*)d_in[12];
  const float* Wh1 = (const float*)d_in[13];  const float* bh1 = (const float*)d_in[14];
  const float* Wv1 = (const float*)d_in[15];  const float* bv1 = (const float*)d_in[16];

  const int total = in_sizes[0] / D_DIM;  // 20480 packed tokens

  // workspace carve-up (256B aligned)
  char* p = (char*)d_ws;
  auto take = [&](size_t bytes) {
    char* r = p;
    p += (bytes + 255) & ~(size_t)255;
    return r;
  };
  __bf16* x2fg = (__bf16*)take((size_t)total * H2_DIM * sizeof(__bf16));
  float*  hs0  = (float*)take((size_t)total * H_DIM * sizeof(float));
  __bf16* wb   = (__bf16*)take((size_t)H2_DIM * H_DIM * sizeof(__bf16));  // weight scratch
  __bf16* whb  = (__bf16*)take((size_t)H2_DIM * H_DIM * sizeof(__bf16));  // Wh for recur
  float*  vbuf = (float*)take((size_t)NR_DIM * H2_DIM * sizeof(float));
  float*  fg   = (float*)take((size_t)B_DIM * H2_DIM * sizeof(float));
  float*  h    = (float*)take((size_t)B_DIM * H_DIM * sizeof(float));
  __bf16* hb   = (__bf16*)take((size_t)B_DIM * H_DIM * sizeof(__bf16));
  int*    bsz  = (int*)take(T_MAX * sizeof(int));
  int*    offs = (int*)take(T_MAX * sizeof(int));
  unsigned* bar = (unsigned*)take(256);

  float* out_packed = (float*)d_out;                       // [total, H]
  float* hn0 = out_packed + (size_t)total * H_DIM;         // [B, H]
  float* hn1 = hn0 + (size_t)B_DIM * H_DIM;                // [B, H]

  hipMemsetAsync(bar, 0, 256, stream);
  setup_offsets_kernel<<<1, 256, 0, stream>>>(lengths, bsz, offs);

  const int GEMM_WGS = 512;
  const int RECUR_WGS = 64;  // fixed: tile map assumes 64 WGs x 8 waves

  // ---- layer 0 ----
  f32_to_bf16_kernel<<<256, 256, 0, stream>>>(Wx0, wb, H2_DIM * D_DIM);
  gemm_wmma_kernel<<<GEMM_WGS, 256, 0, stream>>>(x, wb, bx0, total, H2_DIM, D_DIM,
                                                 nullptr, x2fg);
  f32_to_bf16_kernel<<<256, 256, 0, stream>>>(Wv0, wb, H2_DIM * H_DIM);
  gemm_wmma_kernel<<<32, 256, 0, stream>>>(basis0, wb, bv0, NR_DIM, H2_DIM, H_DIM,
                                           vbuf, nullptr);
  f32_to_bf16_kernel<<<256, 256, 0, stream>>>(Wh0, whb, H2_DIM * H_DIM);
  tpru_recur_kernel<<<RECUR_WGS, 256, 0, stream>>>(x2fg, whb, bh0, vbuf, lengths, bsz,
                                                   offs, h, hb, fg, hs0, hn0, bar);
  // ---- layer 1 ----
  f32_to_bf16_kernel<<<256, 256, 0, stream>>>(Wx1, wb, H2_DIM * H_DIM);
  gemm_wmma_kernel<<<GEMM_WGS, 256, 0, stream>>>(hs0, wb, bx1, total, H2_DIM, H_DIM,
                                                 nullptr, x2fg);
  f32_to_bf16_kernel<<<256, 256, 0, stream>>>(Wv1, wb, H2_DIM * H_DIM);
  gemm_wmma_kernel<<<32, 256, 0, stream>>>(basis1, wb, bv1, NR_DIM, H2_DIM, H_DIM,
                                           vbuf, nullptr);
  f32_to_bf16_kernel<<<256, 256, 0, stream>>>(Wh1, whb, H2_DIM * H_DIM);
  tpru_recur_kernel<<<RECUR_WGS, 256, 0, stream>>>(x2fg, whb, bh1, vbuf, lengths, bsz,
                                                   offs, h, hb, fg, out_packed, hn1, bar);
}
